// MultiHeadAttention_51384988729754
// MI455X (gfx1250) — matrix operand order repaired and checked by reading, not compiled, not run on hardware
//
#include <hip/hip_runtime.h>
#include <cstdint>

// ---------------------------------------------------------------------------
// MI455X (gfx1250) multi-head attention, bf16 WMMA path, flash-style softmax.
// B=2, S=2048, H=1024, NH=16, HD=64. All matmuls via v_wmma_f32_16x16x32_bf16.
// Projections use async global->LDS staging (ASYNCcnt) for the weight tile,
// double-buffered, 32x64 wave tiles (8 WMMAs per K-step per wave).
// Workspace usage: 48 MB of d_ws (bf16 staging for X, Q, K, V^T, ctx, 4x W^T).
// ---------------------------------------------------------------------------

#define BATCH   2
#define SEQ     2048
#define HDIM    1024
#define NHEADS  16
#define HEADD   64
#define MROWS   (BATCH * SEQ)        // 4096

typedef __attribute__((ext_vector_type(16))) __bf16 v16bf;
typedef __attribute__((ext_vector_type(8)))  __bf16 v8bf;
typedef __attribute__((ext_vector_type(8)))  float  v8f;

static __device__ __forceinline__ v8f wmma_bf16(v16bf a, v16bf b, v8f c) {
    return __builtin_amdgcn_wmma_f32_16x16x32_bf16(
        /*neg_a=*/false, a, /*neg_b=*/false, b,
        /*c_mod=*/(short)0, c, /*reuse_a=*/false, /*reuse_b=*/false);
}

static __device__ __forceinline__ float rowmax16(float v) {
    v = fmaxf(v, __shfl_xor(v, 1, 32));
    v = fmaxf(v, __shfl_xor(v, 2, 32));
    v = fmaxf(v, __shfl_xor(v, 4, 32));
    v = fmaxf(v, __shfl_xor(v, 8, 32));
    return v;
}
static __device__ __forceinline__ float rowsum16(float v) {
    v += __shfl_xor(v, 1, 32);
    v += __shfl_xor(v, 2, 32);
    v += __shfl_xor(v, 4, 32);
    v += __shfl_xor(v, 8, 32);
    return v;
}

// Async global->LDS 16-byte copy per lane (ASYNCcnt-tracked, ISA 15.18.3 #98).
static __device__ __forceinline__ void async_g2l_b128(uint32_t lds_off,
                                                      const __bf16* gaddr) {
    asm volatile("global_load_async_to_lds_b128 %0, %1, off"
                 :: "v"(lds_off), "v"(gaddr) : "memory");
}

// ------------------------- fp32 -> bf16 convert ----------------------------
__global__ __launch_bounds__(256)
void cvt_f32_bf16(const float* __restrict__ in, __bf16* __restrict__ out, int n) {
    int i = (blockIdx.x * 256 + threadIdx.x) * 4;
    if (i + 3 < n) {
        float4 v = *(const float4*)(in + i);
        out[i + 0] = (__bf16)v.x;
        out[i + 1] = (__bf16)v.y;
        out[i + 2] = (__bf16)v.z;
        out[i + 3] = (__bf16)v.w;
    }
}

// --------------- W[K][N] fp32 -> Wt[N][K] bf16 (LDS tile transpose) --------
__global__ __launch_bounds__(256)
void transpose_w_bf16(const float* __restrict__ W, __bf16* __restrict__ Wt,
                      int K, int N) {
    __shared__ float tile[32][33];
    int n = blockIdx.x * 32 + threadIdx.x;     // column (out feature)
    int k0 = blockIdx.y * 32;                  // row base (in feature)
#pragma unroll
    for (int j = (int)threadIdx.y; j < 32; j += 8)
        tile[j][threadIdx.x] = W[(size_t)(k0 + j) * N + n];
    __syncthreads();
    int nb = blockIdx.x * 32;
#pragma unroll
    for (int j = (int)threadIdx.y; j < 32; j += 8)
        Wt[(size_t)(nb + j) * K + k0 + threadIdx.x] = (__bf16)tile[threadIdx.x][j];
}

// ----------------------------- WMMA GEMM -----------------------------------
// C[M,N] = A[M,K] (bf16, row-major) x Bt[N,K] (bf16, pre-transposed) + bias.
// Block = 8 waves in a 4x2 grid; wave tile = 32x64 (2 A frags x 4 B frags =
// 8 WMMAs per 32-deep K-step). B tile (128x32) staged into LDS with async
// global->LDS copies, double buffered; A rows stream from L2. B fragments are
// hoisted into registers first so the DS clause is waited on once, then the
// 8 WMMAs issue back-to-back.
// mode 0: bf16 out row-major          (Q, K projections)
// mode 1: bf16 out as V^T [B,NH,HD,S] (V projection)
// mode 2: fp32 out row-major + bias   (output projection -> d_out)
__global__ __launch_bounds__(256)
void gemm_wmma(const __bf16* __restrict__ A, const __bf16* __restrict__ Bt,
               const float* __restrict__ bias, __bf16* __restrict__ outB,
               float* __restrict__ outF, int M, int N, int K, int mode) {
    __shared__ __align__(32) __bf16 Bs[2][128 * 32];   // 2 x 8 KB

    const int tid  = threadIdx.x;
    const int lane = tid & 31;
    const int w    = tid >> 5;
    const int lr   = lane & 15;
    const int hi   = lane >> 4;
    const int m0   = blockIdx.y * 128 + (w >> 1) * 32;
    const int n0l  = (w & 1) * 64;             // n offset inside the LDS tile
    const int nblk = blockIdx.x * 128;

    v8f acc[2][4] = {};

    const __bf16* Ar0 = A + (size_t)(m0 + lr) * K + hi * 16;
    const __bf16* Ar1 = Ar0 + (size_t)16 * K;

    // Stage the 128x32 B tile for k0 into Bs[buf]: 512 16B chunks, 2/thread.
    // chunk c: row = c>>2 (LDS row stride 64B), seg = c&3.
    auto stage = [&](int k0, int buf) {
        uint32_t base = (uint32_t)(uintptr_t)(&Bs[buf][0]);
#pragma unroll
        for (int q = 0; q < 2; ++q) {
            int c = tid + q * 256;
            const __bf16* src =
                Bt + (size_t)(nblk + (c >> 2)) * K + k0 + (c & 3) * 8;
            async_g2l_b128(base + (uint32_t)c * 16u, src);
        }
    };

    stage(0, 0);
    int buf = 0;
    for (int k0 = 0; k0 < K; k0 += 32, buf ^= 1) {
        if (k0 + 32 < K) {
            stage(k0 + 32, buf ^ 1);                       // prefetch next
            asm volatile("s_wait_asynccnt 0x2" ::: "memory");  // cur tile landed
        } else {
            asm volatile("s_wait_asynccnt 0x0" ::: "memory");
        }
        __syncthreads();

        // A fragments (global, L2-resident)
        v16bf a0, a1;
        {
            v8bf l0 = *(const v8bf*)(Ar0 + k0);
            v8bf h0 = *(const v8bf*)(Ar0 + k0 + 8);
            v8bf l1 = *(const v8bf*)(Ar1 + k0);
            v8bf h1 = *(const v8bf*)(Ar1 + k0 + 8);
#pragma unroll
            for (int i = 0; i < 8; ++i) {
                a0[i] = l0[i]; a0[i + 8] = h0[i];
                a1[i] = l1[i]; a1[i + 8] = h1[i];
            }
        }
        // All four B fragments first (one DS clause, one wait) ...
        v16bf b[4];
#pragma unroll
        for (int j = 0; j < 4; ++j)
            b[j] = *(const v16bf*)&Bs[buf][(n0l + j * 16 + lr) * 32 + hi * 16];
        // ... then 8 independent WMMAs back-to-back.
#pragma unroll
        for (int j = 0; j < 4; ++j) {
            acc[0][j] = wmma_bf16(a0, b[j], acc[0][j]);
            acc[1][j] = wmma_bf16(a1, b[j], acc[1][j]);
        }
        __syncthreads();   // all waves done reading before next overwrite
    }

#pragma unroll
    for (int r = 0; r < 2; ++r) {
#pragma unroll
        for (int j = 0; j < 4; ++j) {
            int col = nblk + n0l + j * 16 + lr;
            float bb = bias[col];
#pragma unroll
            for (int i = 0; i < 8; ++i) {
                int row = m0 + r * 16 + hi * 8 + i;
                float v = acc[r][j][i] + bb;
                if (mode == 2) {
                    outF[(size_t)row * N + col] = v;
                } else if (mode == 0) {
                    outB[(size_t)row * N + col] = (__bf16)v;
                } else {  // V^T: [B, NH, HD, S]
                    int h = col >> 6, d = col & 63;
                    int bi = row / SEQ, s = row - bi * SEQ;
                    outB[(((size_t)(bi * NHEADS + h)) * HEADD + d) * SEQ + s] =
                        (__bf16)v;
                }
            }
        }
    }
}

// ------------------------- flash attention ---------------------------------
// One wave handles a 16-query tile of one head. 32 keys per step:
// 4 WMMAs for Q.K^T (K=64 contraction), online softmax, probs staged through
// LDS to convert D-layout -> A-layout, 4 WMMAs for P.V. ctx out as bf16 [BS,H].
__global__ __launch_bounds__(256)
void attn_wmma(const __bf16* __restrict__ Q, const __bf16* __restrict__ Kmat,
               const __bf16* __restrict__ Vt, const float* __restrict__ mask,
               __bf16* __restrict__ ctx) {
    __shared__ __align__(32) __bf16 Plds[8][16 * 32];

    const int lane = threadIdx.x & 31;
    const int w    = threadIdx.x >> 5;
    const int lr   = lane & 15;
    const int hi   = lane >> 4;

    const int gw    = blockIdx.x * 8 + w;
    const int qt    = gw & (SEQ / 16 - 1);       // 128 q-tiles per head
    const int headg = gw >> 7;
    const int h     = headg & (NHEADS - 1);
    const int b     = headg >> 4;
    const int q0    = qt * 16;

    // Q A-fragments: rows q0..q0+15, contraction d = 0..63 (two K=32 frags)
    const __bf16* qbase = Q + (size_t)(b * SEQ + q0 + lr) * HDIM + h * HEADD;
    v16bf aq0, aq1;
    {
        v8bf l0 = *(const v8bf*)(qbase + hi * 16);
        v8bf h0 = *(const v8bf*)(qbase + hi * 16 + 8);
        v8bf l1 = *(const v8bf*)(qbase + 32 + hi * 16);
        v8bf h1 = *(const v8bf*)(qbase + 32 + hi * 16 + 8);
#pragma unroll
        for (int i = 0; i < 8; ++i) {
            aq0[i] = l0[i]; aq0[i + 8] = h0[i];
            aq1[i] = l1[i]; aq1[i + 8] = h1[i];
        }
    }

    const __bf16* kbase = Kmat + (size_t)(b * SEQ) * HDIM + h * HEADD;
    const __bf16* vbase = Vt + (size_t)(b * NHEADS + h) * HEADD * SEQ;
    const float*  mbase = mask + (size_t)b * SEQ;

    float mi[8], li[8];
#pragma unroll
    for (int i = 0; i < 8; ++i) { mi[i] = -3.0e38f; li[i] = 0.0f; }
    v8f acc[4] = {};

    for (int kt = 0; kt < SEQ; kt += 32) {
        // ---- scores: two 16x16 tiles (keys kt..kt+15, kt+16..kt+31)
        const __bf16* kp0 = kbase + (size_t)(kt + lr) * HDIM + hi * 16;
        const __bf16* kp1 = kbase + (size_t)(kt + 16 + lr) * HDIM + hi * 16;
        v16bf bk0a = *(const v16bf*)(kp0);
        v16bf bk0b = *(const v16bf*)(kp0 + 32);
        v16bf bk1a = *(const v16bf*)(kp1);
        v16bf bk1b = *(const v16bf*)(kp1 + 32);
        v8f s0 = {}, s1 = {};
        s0 = wmma_bf16(aq0, bk0a, s0);
        s0 = wmma_bf16(aq1, bk0b, s0);
        s1 = wmma_bf16(aq0, bk1a, s1);
        s1 = wmma_bf16(aq1, bk1b, s1);

        float mk0 = mbase[kt + lr];
        float mk1 = mbase[kt + 16 + lr];

        // ---- online softmax (per lane: 8 rows, col lr / lr+16)
#pragma unroll
        for (int i = 0; i < 8; ++i) {
            float v0 = s0[i] * 0.125f + mk0;     // 1/sqrt(64)
            float v1 = s1[i] * 0.125f + mk1;
            float t  = rowmax16(fmaxf(v0, v1));
            float nm = fmaxf(mi[i], t);
            float alpha = __expf(mi[i] - nm);
            mi[i] = nm;
            float e0 = __expf(v0 - nm);
            float e1 = __expf(v1 - nm);
            li[i] = li[i] * alpha + rowsum16(e0 + e1);
            acc[0][i] *= alpha; acc[1][i] *= alpha;
            acc[2][i] *= alpha; acc[3][i] *= alpha;
            int m = hi * 8 + i;
            Plds[w][m * 32 + lr]      = (__bf16)e0;
            Plds[w][m * 32 + lr + 16] = (__bf16)e1;
        }
        asm volatile("s_wait_dscnt 0x0" ::: "memory");  // wave-local DS ordering

        // ---- reload P as an A-fragment (16x32), then ctx += P.V
        const __bf16* pp = &Plds[w][lr * 32 + hi * 16];
        v8bf plo = *(const v8bf*)pp;
        v8bf phi = *(const v8bf*)(pp + 8);
        v16bf ap;
#pragma unroll
        for (int i = 0; i < 8; ++i) { ap[i] = plo[i]; ap[i + 8] = phi[i]; }
#pragma unroll
        for (int j = 0; j < 4; ++j) {
            const __bf16* vp = vbase + (size_t)(j * 16 + lr) * SEQ + kt + hi * 16;
            v16bf bv = *(const v16bf*)vp;
            acc[j] = wmma_bf16(ap, bv, acc[j]);
        }
    }

    // ---- normalize and write ctx (bf16, [B*S, H])
    __bf16* cb = ctx + (size_t)(b * SEQ + q0) * HDIM + h * HEADD;
#pragma unroll
    for (int i = 0; i < 8; ++i) {
        float inv = 1.0f / li[i];
        int row = hi * 8 + i;
#pragma unroll
        for (int j = 0; j < 4; ++j)
            cb[(size_t)row * HDIM + j * 16 + lr] = (__bf16)(acc[j][i] * inv);
    }
}

// ---------------------------------------------------------------------------
extern "C" void kernel_launch(void* const* d_in, const int* in_sizes, int n_in,
                              void* d_out, int out_size, void* d_ws, size_t ws_size,
                              hipStream_t stream) {
    const float* hs   = (const float*)d_in[0];
    const float* mask = (const float*)d_in[1];
    const float* Wq = (const float*)d_in[2]; const float* bq = (const float*)d_in[3];
    const float* Wk = (const float*)d_in[4]; const float* bk = (const float*)d_in[5];
    const float* Wv = (const float*)d_in[6]; const float* bv = (const float*)d_in[7];
    const float* Wo = (const float*)d_in[8]; const float* bo = (const float*)d_in[9];

    // workspace layout (bf16 elements); total = 24M elems = 48 MB
    const size_t NTOK = (size_t)MROWS * HDIM;      // 4M elements
    __bf16* X   = (__bf16*)d_ws;
    __bf16* Qb  = X   + NTOK;
    __bf16* Kb  = Qb  + NTOK;
    __bf16* Vtb = Kb  + NTOK;
    __bf16* Cb  = Vtb + NTOK;
    __bf16* WtQ = Cb  + NTOK;
    __bf16* WtK = WtQ + (size_t)HDIM * HDIM;
    __bf16* WtV = WtK + (size_t)HDIM * HDIM;
    __bf16* WtO = WtV + (size_t)HDIM * HDIM;

    // 1) convert activations, transpose+convert weights
    cvt_f32_bf16<<<(int)(NTOK / 1024), 256, 0, stream>>>(hs, X, (int)NTOK);
    dim3 tpb(32, 8), tpg(HDIM / 32, HDIM / 32);
    transpose_w_bf16<<<tpg, tpb, 0, stream>>>(Wq, WtQ, HDIM, HDIM);
    transpose_w_bf16<<<tpg, tpb, 0, stream>>>(Wk, WtK, HDIM, HDIM);
    transpose_w_bf16<<<tpg, tpb, 0, stream>>>(Wv, WtV, HDIM, HDIM);
    transpose_w_bf16<<<tpg, tpb, 0, stream>>>(Wo, WtO, HDIM, HDIM);

    // 2) QKV projections (V written transposed for the PV WMMA B-fragments)
    dim3 gg(HDIM / 128, MROWS / 128);
    gemm_wmma<<<gg, 256, 0, stream>>>(X, WtQ, bq, Qb, nullptr, MROWS, HDIM, HDIM, 0);
    gemm_wmma<<<gg, 256, 0, stream>>>(X, WtK, bk, Kb, nullptr, MROWS, HDIM, HDIM, 0);
    gemm_wmma<<<gg, 256, 0, stream>>>(X, WtV, bv, Vtb, nullptr, MROWS, HDIM, HDIM, 1);

    // 3) flash attention: 4096 query tiles, 8 waves/block
    attn_wmma<<<BATCH * NHEADS * (SEQ / 16) / 8, 256, 0, stream>>>(
        Qb, Kb, Vtb, mask, Cb);

    // 4) output projection -> fp32 d_out
    gemm_wmma<<<gg, 256, 0, stream>>>(Cb, WtO, bo, nullptr, (float*)d_out,
                                      MROWS, HDIM, HDIM, 2);
}
